// VanillaRNN_55465207661163
// MI455X (gfx1250) — matrix operand order repaired and checked by reading, not compiled, not run on hardware
//
#include <hip/hip_runtime.h>
#include <hip/hip_bf16.h>
#include <stdint.h>
#include <stddef.h>

#define BATCH   128
#define SEQ     512
#define IN_DIM  512
#define HID     1024
#define OUT_DIM 512

typedef __attribute__((ext_vector_type(16))) __bf16 v16bf;
typedef __attribute__((ext_vector_type(8)))  __bf16 v8bf;
typedef __attribute__((ext_vector_type(8)))  float  v8f;

// ---------- workspace layout (bytes, 256-aligned) ----------
static constexpr size_t OFF_CNT   = 0;                                   // 256 B (barrier counter)
static constexpr size_t OFF_H0    = 256;                                 // 128*1024*2
static constexpr size_t OFF_H1    = OFF_H0 + (size_t)BATCH*HID*2;        // 128*1024*2
static constexpr size_t OFF_WX    = OFF_H1 + (size_t)BATCH*HID*2;        // 1024*512*2
static constexpr size_t OFF_WH    = OFF_WX + (size_t)HID*IN_DIM*2;       // 1024*1024*2
static constexpr size_t OFF_WP    = OFF_WH + (size_t)HID*HID*2;          // 512*1024*2
static constexpr size_t OFF_XBF   = OFF_WP + (size_t)OUT_DIM*HID*2;      // 128*512*512*2
static constexpr size_t OFF_XPROJ = OFF_XBF + (size_t)BATCH*SEQ*IN_DIM*2;// 512*128*1024*4 fp32

// ---------- fragment helpers ----------
static __device__ __forceinline__ v16bf combine16(v8bf lo, v8bf hi) {
  v16bf r;
#pragma unroll
  for (int i = 0; i < 8; ++i) { r[i] = lo[i]; r[i + 8] = hi[i]; }
  return r;
}

static __device__ __forceinline__ v8f zero8() {
  v8f z = {0.f, 0.f, 0.f, 0.f, 0.f, 0.f, 0.f, 0.f};
  return z;
}

// A fragment: 16x32 bf16 tile, rows [row_base..+15] of row-major A (lda elems), K [k..k+31].
// Lane L: m = L&15, hi = L>>4, holds K = {hi*8..+7, 16+hi*8..+7}.
static __device__ __forceinline__ v16bf load_a_frag(const __bf16* A, int row_base,
                                                    size_t lda, int k) {
  int lane = threadIdx.x & 31;
  int m = lane & 15, hi = lane >> 4;
  const __bf16* p = A + (size_t)(row_base + m) * lda + k;
  v8bf lo = *(const v8bf*)(p + hi * 8);
  v8bf hh = *(const v8bf*)(p + 16 + hi * 8);
  return combine16(lo, hh);
}

// B fragment: 32x16 where B[k][n] = W[col_base+n][k], W row-major (ldb elems).
// Lane L: n = L&15, hi = L>>4, holds K = {hi*8..+7, 16+hi*8..+7}.
static __device__ __forceinline__ v16bf load_b_frag(const __bf16* W, int col_base,
                                                    size_t ldb, int k) {
  int lane = threadIdx.x & 31;
  int n = lane & 15, hi = lane >> 4;
  const __bf16* p = W + (size_t)(col_base + n) * ldb + k + hi * 8;
  v8bf lo = *(const v8bf*)(p);
  v8bf hh = *(const v8bf*)(p + 16);
  return combine16(lo, hh);
}

#define WMMA_BF16(A_, B_, C_) \
  __builtin_amdgcn_wmma_f32_16x16x32_bf16(false, (A_), false, (B_), (short)0, (C_), false, false)

// One wave computes a 32x64 strip: C[row_base..+32, col_base..+64] += A * W^T
// Software-pipelined: fragments for k+32 are loaded before the WMMAs of k.
static __device__ __forceinline__ void gemm_wave_32x64(
    const __bf16* __restrict__ A, int row_base, size_t lda,
    const __bf16* __restrict__ W, int col_base, size_t ldb,
    int K, v8f acc[2][4]) {
  v16bf a0 = load_a_frag(A, row_base,      lda, 0);
  v16bf a1 = load_a_frag(A, row_base + 16, lda, 0);
  v16bf b0 = load_b_frag(W, col_base,      ldb, 0);
  v16bf b1 = load_b_frag(W, col_base + 16, ldb, 0);
  v16bf b2 = load_b_frag(W, col_base + 32, ldb, 0);
  v16bf b3 = load_b_frag(W, col_base + 48, ldb, 0);

  for (int k = 32; k < K; k += 32) {
    // prefetch next iteration's fragments (fresh SSA values -> overlap with WMMAs)
    v16bf na0 = load_a_frag(A, row_base,      lda, k);
    v16bf na1 = load_a_frag(A, row_base + 16, lda, k);
    v16bf nb0 = load_b_frag(W, col_base,      ldb, k);
    v16bf nb1 = load_b_frag(W, col_base + 16, ldb, k);
    v16bf nb2 = load_b_frag(W, col_base + 32, ldb, k);
    v16bf nb3 = load_b_frag(W, col_base + 48, ldb, k);

    acc[0][0] = WMMA_BF16(a0, b0, acc[0][0]);
    acc[0][1] = WMMA_BF16(a0, b1, acc[0][1]);
    acc[0][2] = WMMA_BF16(a0, b2, acc[0][2]);
    acc[0][3] = WMMA_BF16(a0, b3, acc[0][3]);
    acc[1][0] = WMMA_BF16(a1, b0, acc[1][0]);
    acc[1][1] = WMMA_BF16(a1, b1, acc[1][1]);
    acc[1][2] = WMMA_BF16(a1, b2, acc[1][2]);
    acc[1][3] = WMMA_BF16(a1, b3, acc[1][3]);

    a0 = na0; a1 = na1; b0 = nb0; b1 = nb1; b2 = nb2; b3 = nb3;
  }

  acc[0][0] = WMMA_BF16(a0, b0, acc[0][0]);
  acc[0][1] = WMMA_BF16(a0, b1, acc[0][1]);
  acc[0][2] = WMMA_BF16(a0, b2, acc[0][2]);
  acc[0][3] = WMMA_BF16(a0, b3, acc[0][3]);
  acc[1][0] = WMMA_BF16(a1, b0, acc[1][0]);
  acc[1][1] = WMMA_BF16(a1, b1, acc[1][1]);
  acc[1][2] = WMMA_BF16(a1, b2, acc[1][2]);
  acc[1][3] = WMMA_BF16(a1, b3, acc[1][3]);
}

// ---------- kernels ----------
__global__ void k_cvt_bf16(const float* __restrict__ src, __bf16* __restrict__ dst, int n) {
  int i = blockIdx.x * blockDim.x + threadIdx.x;
  int stride = gridDim.x * blockDim.x;
  for (; i < n; i += stride) dst[i] = (__bf16)src[i];
}

// Block = 8 waves = 64 rows x 256 cols; wave w: rows +(w&1)*32, cols +(w>>1)*64.

// xproj[s][b][h] = sum_i x[b][s][i] * Wx[h][i] + Wx_b[h]
// grid: SEQ*8 blocks; blk = (s<<3) | (rb<<2) | cb
__global__ __launch_bounds__(256) void k_xproj(const __bf16* __restrict__ xbf,
                                               const __bf16* __restrict__ Wxbf,
                                               const float* __restrict__ Wxb,
                                               float* __restrict__ xproj) {
  int blk = blockIdx.x;
  int s  = blk >> 3;
  int rb = (blk >> 2) & 1;
  int cb = blk & 3;
  int w = threadIdx.x >> 5;
  int row_base = rb * 64 + (w & 1) * 32;
  int col_base = cb * 256 + (w >> 1) * 64;

  v8f acc[2][4];
#pragma unroll
  for (int mi = 0; mi < 2; ++mi)
#pragma unroll
    for (int j = 0; j < 4; ++j) acc[mi][j] = zero8();

  const __bf16* A = xbf + (size_t)s * IN_DIM;            // row stride SEQ*IN_DIM
  gemm_wave_32x64(A, row_base, (size_t)SEQ * IN_DIM, Wxbf, col_base, IN_DIM, IN_DIM, acc);

  int lane = threadIdx.x & 31;
  int n = lane & 15, hi = lane >> 4;
  float* outp = xproj + (size_t)s * BATCH * HID;
#pragma unroll
  for (int mi = 0; mi < 2; ++mi) {
#pragma unroll
    for (int j = 0; j < 4; ++j) {
      int col = col_base + j * 16 + n;
      float bias = Wxb[col];
#pragma unroll
      for (int r = 0; r < 8; ++r) {
        int row = row_base + mi * 16 + hi * 8 + r;
        outp[(size_t)row * HID + col] = acc[mi][j][r] + bias;
      }
    }
  }
}

// Persistent recurrent kernel: 8 blocks, grid barrier between timesteps.
// h_new = tanh(xproj[t] + h * Wh^T), double-buffered bf16 h in L2-resident global.
__global__ __launch_bounds__(256) void k_rnn(const float* __restrict__ xproj,
                                             const __bf16* __restrict__ Whbf,
                                             __bf16* h0, __bf16* h1,
                                             int* cnt) {
  int wg = blockIdx.x;            // 8 blocks
  int rb = wg & 1;                // 2 row blocks of 64
  int cb = wg >> 1;               // 4 col blocks of 256
  int w = threadIdx.x >> 5;
  int row_base = rb * 64 + (w & 1) * 32;
  int col_base = cb * 256 + (w >> 1) * 64;
  int lane = threadIdx.x & 31;
  int n = lane & 15, hi = lane >> 4;
  int nblk = (int)gridDim.x;

  __bf16* bufs[2] = { h0, h1 };

  for (int t = 0; t < SEQ; ++t) {
    const __bf16* hc = bufs[t & 1];
    __bf16* hn = bufs[(t + 1) & 1];

    v8f acc[2][4];
#pragma unroll
    for (int mi = 0; mi < 2; ++mi)
#pragma unroll
      for (int j = 0; j < 4; ++j) acc[mi][j] = zero8();

    gemm_wave_32x64(hc, row_base, HID, Whbf, col_base, HID, HID, acc);

    const float* xp = xproj + (size_t)t * BATCH * HID;
#pragma unroll
    for (int mi = 0; mi < 2; ++mi) {
#pragma unroll
      for (int j = 0; j < 4; ++j) {
        int col = col_base + j * 16 + n;
#pragma unroll
        for (int r = 0; r < 8; ++r) {
          int row = row_base + mi * 16 + hi * 8 + r;
          float v = tanhf(acc[mi][j][r] + xp[(size_t)row * HID + col]);
          hn[(size_t)row * HID + col] = (__bf16)v;
        }
      }
    }

    // ---- grid barrier (monotonic counter; all 8 blocks co-resident) ----
    __syncthreads();
    if (threadIdx.x == 0) {
      __hip_atomic_fetch_add(cnt, 1, __ATOMIC_RELEASE, __HIP_MEMORY_SCOPE_AGENT);
      int target = (t + 1) * nblk;
      while (__hip_atomic_load(cnt, __ATOMIC_ACQUIRE, __HIP_MEMORY_SCOPE_AGENT) < target) {
        __builtin_amdgcn_s_sleep(1);
      }
    }
    __syncthreads();
  }
}

// out[b][o] = sum_k hT[b][k] * Wp[o][k] + Wp_b[o]
__global__ __launch_bounds__(256) void k_proj(const __bf16* __restrict__ hT,
                                              const __bf16* __restrict__ Wpbf,
                                              const float* __restrict__ Wpb,
                                              float* __restrict__ out) {
  int wg = blockIdx.x;            // 4 blocks
  int rb = wg & 1;                // 2 row blocks of 64
  int cb = wg >> 1;               // 2 col blocks of 256
  int w = threadIdx.x >> 5;
  int row_base = rb * 64 + (w & 1) * 32;
  int col_base = cb * 256 + (w >> 1) * 64;

  v8f acc[2][4];
#pragma unroll
  for (int mi = 0; mi < 2; ++mi)
#pragma unroll
    for (int j = 0; j < 4; ++j) acc[mi][j] = zero8();

  gemm_wave_32x64(hT, row_base, HID, Wpbf, col_base, HID, HID, acc);

  int lane = threadIdx.x & 31;
  int n = lane & 15, hi = lane >> 4;
#pragma unroll
  for (int mi = 0; mi < 2; ++mi) {
#pragma unroll
    for (int j = 0; j < 4; ++j) {
      int col = col_base + j * 16 + n;
      float bias = Wpb[col];
#pragma unroll
      for (int r = 0; r < 8; ++r) {
        int row = row_base + mi * 16 + hi * 8 + r;
        out[(size_t)row * OUT_DIM + col] = acc[mi][j][r] + bias;
      }
    }
  }
}

// ---------- launch ----------
extern "C" void kernel_launch(void* const* d_in, const int* in_sizes, int n_in,
                              void* d_out, int out_size, void* d_ws, size_t ws_size,
                              hipStream_t stream) {
  const float* x    = (const float*)d_in[0];
  const float* Wx_w = (const float*)d_in[1];
  const float* Wx_b = (const float*)d_in[2];
  const float* Wh_w = (const float*)d_in[3];
  const float* Wp_w = (const float*)d_in[4];
  const float* Wp_b = (const float*)d_in[5];
  float* out = (float*)d_out;

  char* ws = (char*)d_ws;
  int*    cnt   = (int*)   (ws + OFF_CNT);
  __bf16* h0    = (__bf16*)(ws + OFF_H0);
  __bf16* h1    = (__bf16*)(ws + OFF_H1);
  __bf16* Wxbf  = (__bf16*)(ws + OFF_WX);
  __bf16* Whbf  = (__bf16*)(ws + OFF_WH);
  __bf16* Wpbf  = (__bf16*)(ws + OFF_WP);
  __bf16* xbf   = (__bf16*)(ws + OFF_XBF);
  float*  xproj = (float*) (ws + OFF_XPROJ);

  hipMemsetAsync(cnt, 0, 256, stream);
  hipMemsetAsync(h0, 0, (size_t)BATCH * HID * 2, stream);

  k_cvt_bf16<<<4096, 256, 0, stream>>>(x,    xbf,  BATCH * SEQ * IN_DIM);
  k_cvt_bf16<<<512,  256, 0, stream>>>(Wx_w, Wxbf, HID * IN_DIM);
  k_cvt_bf16<<<1024, 256, 0, stream>>>(Wh_w, Whbf, HID * HID);
  k_cvt_bf16<<<512,  256, 0, stream>>>(Wp_w, Wpbf, OUT_DIM * HID);

  k_xproj<<<SEQ * 8, 256, 0, stream>>>(xbf, Wxbf, Wx_b, xproj);
  k_rnn  <<<8,       256, 0, stream>>>(xproj, Whbf, h0, h1, cnt);
  k_proj <<<4,       256, 0, stream>>>(h0, Wpbf, Wp_b, out);
}
